// StructureAwareGraphEmbedding_50629074485937
// MI455X (gfx1250) — matrix operand order repaired and checked by reading, not compiled, not run on hardware
//
#include <hip/hip_runtime.h>
#include <hip/hip_bf16.h>

// ---------------------------------------------------------------------------
// Types for CDNA5 WMMA (wave32)
// ---------------------------------------------------------------------------
typedef __attribute__((ext_vector_type(16))) _Float16 v16h;
typedef __attribute__((ext_vector_type(8)))  float    v8f;
typedef __attribute__((ext_vector_type(8)))  _Float16 h8v;

#define N_NODES 2048
#define SEQ     64
#define D       256
#define MROWS   (N_NODES * SEQ)   // 131072
#define NH      8
#define DH      32
#define GH      4
#define NEDGES  32768
#define HID     512

__device__ inline v8f vzero8() {
  v8f z;
#pragma unroll
  for (int i = 0; i < 8; ++i) z[i] = 0.0f;
  return z;
}

__device__ inline v16h cat8(h8v a, h8v b) {
  return __builtin_shufflevector(a, b, 0, 1, 2, 3, 4, 5, 6, 7,
                                 8, 9, 10, 11, 12, 13, 14, 15);
}

__device__ inline void atomicMaxF(float* addr, float val) {
  unsigned int* ua = (unsigned int*)addr;
  unsigned int old = *ua;
  while (__uint_as_float(old) < val) {
    unsigned int assumed = old;
    old = atomicCAS(ua, assumed, __float_as_uint(val));
    if (old == assumed) break;
  }
}

// ---------------------------------------------------------------------------
// WMMA GEMM: C[M,N] = A[M,K](f16) x B[K,N](f16) (+bias) (+relu)
// Block = 128 threads (4 waves), tile 64x128, K-step 32, software-pipelined
// staging: next K-tile global loads overlap current tile's 8 WMMAs.
// A staged row-major [m][k]; B staged TRANSPOSED [n][k] so every fragment is
// two contiguous 16B runs -> ds_load_b128 pairs.
// M%64==0, N%128==0, K%32==0 (guaranteed by the model shapes).
// ---------------------------------------------------------------------------
__global__ __launch_bounds__(128)
void gemm_wmma_kernel(const _Float16* __restrict__ A, const _Float16* __restrict__ B,
                      const float* __restrict__ bias, float* __restrict__ C,
                      _Float16* __restrict__ C16, int M, int N, int K, int relu)
{
  __shared__ _Float16 As[64][40];    // [m][k], padded
  __shared__ _Float16 Bt[128][40];   // [n][k], padded (transposed B tile)

  const int tid  = threadIdx.x;
  const int lane = tid & 31;
  const int wave = tid >> 5;                 // 0..3 -> rows 16*wave..
  const int bm   = blockIdx.y * 64;
  const int bn   = blockIdx.x * 128;

  const int am    = lane & 15;               // A row within 16
  const int kAoff = (lane & 16);             // 0 or 16 (A frag k base)
  const int nl    = lane & 15;               // B col within 16
  const int kBoff = (lane & 16);             // 0 or 16 (B frag k base)

  // staging coordinates
  const int ar = tid >> 1;                   // A row 0..63
  const int ac = (tid & 1) * 16;             // A col base 0/16
  const int br = tid >> 2;                   // B k-row 0..31
  const int bc = (tid & 3) * 32;             // B n base 0/32/64/96

  v8f acc[8];
#pragma unroll
  for (int i = 0; i < 8; ++i) acc[i] = vzero8();

  h8v a_reg[2], b_reg[4];
  auto load_tile = [&](int k0) {
    const h8v* srcA = (const h8v*)(A + (size_t)(bm + ar) * K + k0 + ac);
    a_reg[0] = srcA[0];
    a_reg[1] = srcA[1];
    const h8v* srcB = (const h8v*)(B + (size_t)(k0 + br) * N + bn + bc);
#pragma unroll
    for (int v = 0; v < 4; ++v) b_reg[v] = srcB[v];
  };

  load_tile(0);

  for (int k0 = 0; k0 < K; k0 += 32) {
    // commit staged registers to LDS
    *(h8v*)&As[ar][ac]     = a_reg[0];
    *(h8v*)&As[ar][ac + 8] = a_reg[1];
#pragma unroll
    for (int v = 0; v < 4; ++v) {
#pragma unroll
      for (int i = 0; i < 8; ++i) Bt[bc + v * 8 + i][br] = b_reg[v][i];
    }
    __syncthreads();

    // overlap: issue next tile's global loads while this tile computes
    if (k0 + 32 < K) {
      load_tile(k0 + 32);
      __builtin_prefetch(A + (size_t)(bm + ar) * K + k0 + 64, 0, 0);
      __builtin_prefetch(B + (size_t)(k0 + 64 + br) * N + bn, 0, 0);
    }

    // A fragment: two contiguous 8-half runs -> ds_load_b128 x2
    const int arow = wave * 16 + am;
    const h8v a0 = *(const h8v*)&As[arow][kAoff];
    const h8v a1 = *(const h8v*)&As[arow][kAoff + 8];
    const v16h afrag = cat8(a0, a1);

#pragma unroll
    for (int nt = 0; nt < 8; ++nt) {
      const int brow = nt * 16 + nl;
      const h8v b0 = *(const h8v*)&Bt[brow][kBoff];
      const h8v b1 = *(const h8v*)&Bt[brow][kBoff + 8];
      const v16h bfrag = cat8(b0, b1);
      acc[nt] = __builtin_amdgcn_wmma_f32_16x16x32_f16(false, afrag, false, bfrag,
                                                       (short)0, acc[nt], false, false);
    }
    __syncthreads();
  }

  // epilogue
#pragma unroll
  for (int nt = 0; nt < 8; ++nt) {
    const int col = bn + nt * 16 + nl;
    const float bv = bias ? bias[col] : 0.0f;
#pragma unroll
    for (int r = 0; r < 8; ++r) {
      const int row = bm + wave * 16 + ((lane & 16) >> 1) + r;
      float v = acc[nt][r] + bv;
      if (relu) v = v > 0.0f ? v : 0.0f;
      if (C)   C[(size_t)row * N + col]   = v;
      if (C16) C16[(size_t)row * N + col] = (_Float16)v;
    }
  }
}

// ---------------------------------------------------------------------------
// Fused attention: one block per (node, head). S = Q K^T, softmax, O = P V.
// qkv: [MROWS, 768] f16 (q|k|v, each 256 = 8 heads x 32). out: [MROWS,256] f16
// Q,K row-major (K row-major == K^T pre-transposed for B frags);
// V stored transposed Vt[d][seq] so P@V B-frags are contiguous too.
// ---------------------------------------------------------------------------
__global__ __launch_bounds__(128)
void attention_kernel(const _Float16* __restrict__ qkv, _Float16* __restrict__ out)
{
  __shared__ _Float16 Qs[64][40];   // [seq][dh]
  __shared__ _Float16 Ks[64][40];   // [seq][dh] (== K^T in [n][k] form)
  __shared__ _Float16 Vt[32][72];   // [dh][seq] transposed V
  __shared__ float    Ss[64][68];   // scores
  __shared__ _Float16 Ps[64][72];   // softmaxed probs, row-major

  const int tid  = threadIdx.x;
  const int lane = tid & 31;
  const int wave = tid >> 5;
  const int node = blockIdx.x >> 3;
  const int head = blockIdx.x & 7;
  const size_t rowbase = (size_t)node * SEQ;
  const int qc = head * DH, kc = D + head * DH, vc = 2 * D + head * DH;

#pragma unroll
  for (int i = 0; i < 2; ++i) {
    const int v = tid + i * 128;          // 0..255
    const int r = v >> 2;                 // seq row
    const int c = (v & 3) * 8;            // 0/8/16/24
    const _Float16* base = qkv + (rowbase + r) * (3 * D);
    *(h8v*)&Qs[r][c] = *(const h8v*)(base + qc + c);
    *(h8v*)&Ks[r][c] = *(const h8v*)(base + kc + c);
    const h8v vv = *(const h8v*)(base + vc + c);
#pragma unroll
    for (int j = 0; j < 8; ++j) Vt[c + j][r] = vv[j];   // transpose scatter
  }
  __syncthreads();

  const int am    = lane & 15;
  const int kAoff = (lane & 16);
  const int nl    = lane & 15;
  const int kBoff = (lane & 16);

  // S = Q @ K^T (K-dim = 32, one wmma per 16x16 tile)
  const h8v q0 = *(const h8v*)&Qs[wave * 16 + am][kAoff];
  const h8v q1 = *(const h8v*)&Qs[wave * 16 + am][kAoff + 8];
  const v16h aq = cat8(q0, q1);
#pragma unroll
  for (int nt = 0; nt < 4; ++nt) {
    const h8v k0v = *(const h8v*)&Ks[nt * 16 + nl][kBoff];
    const h8v k1v = *(const h8v*)&Ks[nt * 16 + nl][kBoff + 8];
    const v16h bk = cat8(k0v, k1v);
    v8f s = vzero8();
    s = __builtin_amdgcn_wmma_f32_16x16x32_f16(false, aq, false, bk, (short)0, s, false, false);
#pragma unroll
    for (int r = 0; r < 8; ++r) {
      const int row = wave * 16 + ((lane & 16) >> 1) + r;
      Ss[row][nt * 16 + nl] = s[r];
    }
  }
  __syncthreads();

  // softmax rows (scale 1/sqrt(DH) folded into exponent)
  if (tid < 64) {
    float mx = -3.0e38f;
#pragma unroll
    for (int j = 0; j < 64; ++j) mx = fmaxf(mx, Ss[tid][j]);
    const float invs = 0.17677669529663687f;  // 1/sqrt(32)
    float sum = 0.0f;
#pragma unroll
    for (int j = 0; j < 64; ++j) {
      const float e = __expf((Ss[tid][j] - mx) * invs);
      sum += e;
      Ss[tid][j] = e;
    }
    const float inv = 1.0f / sum;
#pragma unroll
    for (int j = 0; j < 64; ++j) Ps[tid][j] = (_Float16)(Ss[tid][j] * inv);
  }
  __syncthreads();

  // O = P @ V (K-dim 64 = 2 wmma steps, N = 32 = 2 tiles)
#pragma unroll
  for (int nt = 0; nt < 2; ++nt) {
    v8f o = vzero8();
#pragma unroll
    for (int kk0 = 0; kk0 < 2; ++kk0) {
      const h8v p0 = *(const h8v*)&Ps[wave * 16 + am][kk0 * 32 + kAoff];
      const h8v p1 = *(const h8v*)&Ps[wave * 16 + am][kk0 * 32 + kAoff + 8];
      const v16h ap = cat8(p0, p1);
      const h8v v0 = *(const h8v*)&Vt[nt * 16 + nl][kk0 * 32 + kBoff];
      const h8v v1 = *(const h8v*)&Vt[nt * 16 + nl][kk0 * 32 + kBoff + 8];
      const v16h bv = cat8(v0, v1);
      o = __builtin_amdgcn_wmma_f32_16x16x32_f16(false, ap, false, bv, (short)0, o, false, false);
    }
#pragma unroll
    for (int r = 0; r < 8; ++r) {
      const int row = wave * 16 + ((lane & 16) >> 1) + r;
      const int col = head * DH + nt * 16 + nl;
      out[(rowbase + row) * D + col] = (_Float16)o[r];
    }
  }
}

// ---------------------------------------------------------------------------
// Elementwise / reduction kernels
// ---------------------------------------------------------------------------
__global__ void f32_to_f16_kernel(const float* __restrict__ src, _Float16* __restrict__ dst, int n)
{
  const int i = blockIdx.x * blockDim.x + threadIdx.x;
  if (i < n) dst[i] = (_Float16)src[i];
}

__global__ void zero_kernel(float* __restrict__ p, int n)
{
  const int i = blockIdx.x * blockDim.x + threadIdx.x;
  if (i < n) p[i] = 0.0f;
}

__global__ __launch_bounds__(256)
void embed_kernel(const float* __restrict__ tok_emb, const int* __restrict__ toks,
                  float* __restrict__ x32, _Float16* __restrict__ x16)
{
  const int row = blockIdx.x, d = threadIdx.x;
  const int s = row & (SEQ - 1);
  const int tok = toks[row];
  float v = tok_emb[(size_t)tok * D + d] * 16.0f;       // sqrt(256)
  const int p = d >> 1;
  const float freq = __expf((float)(2 * p) * (-9.210340371976184f / 256.0f));
  const float ang = (float)s * freq;
  v += (d & 1) ? __cosf(ang) : __sinf(ang);
  x32[(size_t)row * D + d] = v;
  x16[(size_t)row * D + d] = (_Float16)v;
}

__global__ __launch_bounds__(256)
void ln_residual_kernel(float* __restrict__ x32, const float* __restrict__ add,
                        const float* __restrict__ g, const float* __restrict__ b,
                        _Float16* __restrict__ x16)
{
  __shared__ float red[256];
  const int row = blockIdx.x, d = threadIdx.x;
  const size_t idx = (size_t)row * D + d;
  const float v = x32[idx] + add[idx];
  red[d] = v;
  __syncthreads();
  for (int s = 128; s > 0; s >>= 1) { if (d < s) red[d] += red[d + s]; __syncthreads(); }
  const float mean = red[0] / 256.0f;
  __syncthreads();
  const float c = v - mean;
  red[d] = c * c;
  __syncthreads();
  for (int s = 128; s > 0; s >>= 1) { if (d < s) red[d] += red[d + s]; __syncthreads(); }
  const float var = red[0] / 256.0f;
  const float o = c * rsqrtf(var + 1e-5f) * g[d] + b[d];
  x32[idx] = o;
  x16[idx] = (_Float16)o;
}

__global__ __launch_bounds__(256)
void pool_kernel(const float* __restrict__ x32, const int* __restrict__ types,
                 const float* __restrict__ type_emb, float* __restrict__ h32,
                 _Float16* __restrict__ h16)
{
  const int n = blockIdx.x, d = threadIdx.x;
  float s = 0.0f;
  for (int t = 0; t < SEQ; ++t) s += x32[((size_t)n * SEQ + t) * D + d];
  const float v = s * (1.0f / 64.0f) + type_emb[(size_t)types[n] * D + d];
  h32[(size_t)n * D + d] = v;
  h16[(size_t)n * D + d] = (_Float16)v;
}

__global__ __launch_bounds__(128)
void gat_scores_kernel(const float* __restrict__ xg, const float* __restrict__ asrc,
                       const float* __restrict__ adst, float* __restrict__ a_s,
                       float* __restrict__ a_d)
{
  const int n = blockIdx.x, g = threadIdx.x >> 5, lane = threadIdx.x & 31;
  const float* xrow = xg + ((size_t)n * GH + g) * D;
  const float* as = asrc + g * D;
  const float* ad = adst + g * D;
  float ss = 0.0f, sd = 0.0f;
  for (int i = lane; i < D; i += 32) { const float x = xrow[i]; ss += x * as[i]; sd += x * ad[i]; }
  for (int o = 16; o > 0; o >>= 1) { ss += __shfl_down(ss, o); sd += __shfl_down(sd, o); }
  if (lane == 0) { a_s[n * GH + g] = ss; a_d[n * GH + g] = sd; }
}

__global__ void gat_init_m_kernel(const float* __restrict__ a_s, const float* __restrict__ a_d,
                                  float* __restrict__ eself, float* __restrict__ m, int n4)
{
  const int i = blockIdx.x * blockDim.x + threadIdx.x;
  if (i >= n4) return;
  float v = a_s[i] + a_d[i];
  v = v > 0.0f ? v : 0.2f * v;
  eself[i] = v;
  m[i] = v;
}

__global__ void gat_edge_max_kernel(const int* __restrict__ src, const int* __restrict__ dst,
                                    const int* __restrict__ et, int r,
                                    const float* __restrict__ a_s, const float* __restrict__ a_d,
                                    float* __restrict__ m, int E)
{
  const int e = blockIdx.x * blockDim.x + threadIdx.x;
  if (e >= E || et[e] != r) return;
  const int s = src[e], d = dst[e];
#pragma unroll
  for (int g = 0; g < GH; ++g) {
    float v = a_s[s * GH + g] + a_d[d * GH + g];
    v = v > 0.0f ? v : 0.2f * v;
    atomicMaxF(&m[d * GH + g], v);
  }
}

__global__ __launch_bounds__(128)
void gat_edge_accum_kernel(const int* __restrict__ src, const int* __restrict__ dst,
                           const int* __restrict__ et, int r,
                           const float* __restrict__ a_s, const float* __restrict__ a_d,
                           const float* __restrict__ m, const float* __restrict__ xg,
                           float* __restrict__ denom, float* __restrict__ num)
{
  const int e = blockIdx.x;
  if (et[e] != r) return;
  const int g = threadIdx.x >> 5, lane = threadIdx.x & 31;
  const int s = src[e], d = dst[e];
  float v = a_s[s * GH + g] + a_d[d * GH + g];
  v = v > 0.0f ? v : 0.2f * v;
  const float w = __expf(v - m[d * GH + g]);
  if (lane == 0) atomicAdd(&denom[d * GH + g], w);
  const float* xrow = xg + ((size_t)s * GH + g) * D;
  float* nrow = num + ((size_t)d * GH + g) * D;
  for (int i = lane; i < D; i += 32) atomicAdd(&nrow[i], w * xrow[i]);
}

__global__ __launch_bounds__(256)
void gat_self_finish_kernel(const float* __restrict__ eself, const float* __restrict__ m,
                            const float* __restrict__ denom, const float* __restrict__ num,
                            const float* __restrict__ xg, const float* __restrict__ bias,
                            float* __restrict__ h_new)
{
  const int n = blockIdx.x, d = threadIdx.x;
  float acc = 0.0f;
#pragma unroll
  for (int g = 0; g < GH; ++g) {
    const float ws = __expf(eself[n * GH + g] - m[n * GH + g]);
    const float den = denom[n * GH + g] + ws;
    acc += (num[((size_t)n * GH + g) * D + d] + ws * xg[((size_t)n * GH + g) * D + d]) / den;
  }
  h_new[(size_t)n * D + d] += acc * 0.25f + bias[d];
}

__global__ __launch_bounds__(256)
void relu_residual_kernel(float* __restrict__ h32, const float* __restrict__ h_new,
                          _Float16* __restrict__ h16)
{
  const int n = blockIdx.x, d = threadIdx.x;
  const size_t i = (size_t)n * D + d;
  float v = h32[i] + h_new[i];
  v = v > 0.0f ? v : 0.0f;
  h32[i] = v;
  h16[i] = (_Float16)v;
}

__global__ void tanh_scale_kernel(float* __restrict__ st, const float* __restrict__ log_c, int ntot)
{
  const int i = blockIdx.x * blockDim.x + threadIdx.x;
  if (i >= ntot) return;
  const float s = sqrtf(__expf(log_c[0]));
  st[i] = tanhf(st[i]) * 2.0f * s;
}

__global__ __launch_bounds__(256)
void lorentz_kernel(const float* __restrict__ st, float* __restrict__ out)
{
  __shared__ float red[256];
  const int n = blockIdx.x, d = threadIdx.x;
  const float v = st[(size_t)n * D + d];
  red[d] = v * v;
  __syncthreads();
  for (int s = 128; s > 0; s >>= 1) { if (d < s) red[d] += red[d + s]; __syncthreads(); }
  float r = sqrtf(red[0]);
  r = fmaxf(r, 1e-8f);
  const float sc = sinhf(r) / r;
  if (d == 0) out[(size_t)n * (D + 1)] = coshf(r);
  out[(size_t)n * (D + 1) + 1 + d] = sc * v;
}

// ---------------------------------------------------------------------------
// Host orchestration
// ---------------------------------------------------------------------------
extern "C" void kernel_launch(void* const* d_in, const int* in_sizes, int n_in,
                              void* d_out, int out_size, void* d_ws, size_t ws_size,
                              hipStream_t stream) {
  (void)in_sizes; (void)n_in; (void)out_size; (void)ws_size;

  const float* tok_emb   = (const float*)d_in[0];
  const float* enc_Wqkv  = (const float*)d_in[1];
  const float* enc_bqkv  = (const float*)d_in[2];
  const float* enc_Wo    = (const float*)d_in[3];
  const float* enc_bo    = (const float*)d_in[4];
  const float* enc_ln1_g = (const float*)d_in[5];
  const float* enc_ln1_b = (const float*)d_in[6];
  const float* enc_ln2_g = (const float*)d_in[7];
  const float* enc_ln2_b = (const float*)d_in[8];
  const float* enc_W1    = (const float*)d_in[9];
  const float* enc_b1    = (const float*)d_in[10];
  const float* enc_W2    = (const float*)d_in[11];
  const float* enc_b2    = (const float*)d_in[12];
  const float* type_emb  = (const float*)d_in[13];
  const float* msg_W     = (const float*)d_in[14];
  const float* msg_b     = (const float*)d_in[15];
  const float* gat_W     = (const float*)d_in[16];
  const float* gat_b     = (const float*)d_in[17];
  const float* gat_asrc  = (const float*)d_in[18];
  const float* gat_adst  = (const float*)d_in[19];
  const float* sp_W1     = (const float*)d_in[20];
  const float* sp_b1     = (const float*)d_in[21];
  const float* sp_W2     = (const float*)d_in[22];
  const float* sp_b2     = (const float*)d_in[23];
  const float* log_c     = (const float*)d_in[24];
  const int* rule_toks   = (const int*)d_in[25];
  const int* rule_types  = (const int*)d_in[26];
  const int* edge_index  = (const int*)d_in[27];
  const int* edge_type   = (const int*)d_in[28];
  const int* e_src = edge_index;
  const int* e_dst = edge_index + NEDGES;

  char* ws = (char*)d_ws;
  size_t off = 0;
  auto alloc = [&](size_t bytes) -> char* {
    char* p = ws + off;
    off += (bytes + 255) & ~(size_t)255;
    return p;
  };

  // f16 weight copies
  _Float16* wqkv16 = (_Float16*)alloc((size_t)2 * D * 3 * D * 2);
  _Float16* wo16   = (_Float16*)alloc((size_t)2 * D * D * 2);
  _Float16* w1_16  = (_Float16*)alloc((size_t)2 * D * D * 2);
  _Float16* w2_16  = (_Float16*)alloc((size_t)2 * D * D * 2);
  _Float16* msg16  = (_Float16*)alloc((size_t)3 * D * D * 2);
  _Float16* gatw16 = (_Float16*)alloc((size_t)4 * D * GH * D * 2);
  _Float16* sp1_16 = (_Float16*)alloc((size_t)D * HID * 2);
  _Float16* sp2_16 = (_Float16*)alloc((size_t)HID * D * 2);
  // activations
  float*    x32    = (float*)   alloc((size_t)MROWS * D * 4);
  _Float16* x16    = (_Float16*)alloc((size_t)MROWS * D * 2);
  _Float16* qkv16  = (_Float16*)alloc((size_t)MROWS * 3 * D * 2);
  _Float16* t16    = (_Float16*)alloc((size_t)MROWS * D * 2);
  float*    t32    = (float*)   alloc((size_t)MROWS * D * 4);
  // node-level buffers
  float*    h32    = (float*)   alloc((size_t)N_NODES * D * 4);
  _Float16* h16    = (_Float16*)alloc((size_t)N_NODES * D * 2);
  float*    hn32   = (float*)   alloc((size_t)N_NODES * D * 4);
  _Float16* th16   = (_Float16*)alloc((size_t)N_NODES * D * 2);
  float*    xg32   = (float*)   alloc((size_t)N_NODES * GH * D * 4);
  float*    a_s    = (float*)   alloc((size_t)N_NODES * GH * 4);
  float*    a_d    = (float*)   alloc((size_t)N_NODES * GH * 4);
  float*    eself  = (float*)   alloc((size_t)N_NODES * GH * 4);
  float*    mbuf   = (float*)   alloc((size_t)N_NODES * GH * 4);
  float*    denom  = (float*)   alloc((size_t)N_NODES * GH * 4);
  float*    num32  = (float*)   alloc((size_t)N_NODES * GH * D * 4);
  _Float16* hid16  = (_Float16*)alloc((size_t)N_NODES * HID * 2);
  float*    st32   = (float*)   alloc((size_t)N_NODES * D * 4);

  auto cvt = [&](const float* s, _Float16* dtp, int n) {
    f32_to_f16_kernel<<<(n + 255) / 256, 256, 0, stream>>>(s, dtp, n);
  };
  cvt(enc_Wqkv, wqkv16, 2 * D * 3 * D);
  cvt(enc_Wo,   wo16,   2 * D * D);
  cvt(enc_W1,   w1_16,  2 * D * D);
  cvt(enc_W2,   w2_16,  2 * D * D);
  cvt(msg_W,    msg16,  3 * D * D);
  cvt(gat_W,    gatw16, 4 * D * GH * D);
  cvt(sp_W1,    sp1_16, D * HID);
  cvt(sp_W2,    sp2_16, HID * D);

  auto gemm = [&](const _Float16* A, const _Float16* B, const float* bias,
                  float* C, _Float16* C16, int M_, int N_, int K_, int relu) {
    dim3 grid(N_ / 128, M_ / 64);
    gemm_wmma_kernel<<<grid, 128, 0, stream>>>(A, B, bias, C, C16, M_, N_, K_, relu);
  };

  // ---- logic encoder --------------------------------------------------------
  embed_kernel<<<MROWS, 256, 0, stream>>>(tok_emb, rule_toks, x32, x16);

  for (int l = 0; l < 2; ++l) {
    gemm(x16, wqkv16 + (size_t)l * D * 3 * D, enc_bqkv + l * 3 * D,
         nullptr, qkv16, MROWS, 3 * D, D, 0);
    attention_kernel<<<N_NODES * NH, 128, 0, stream>>>(qkv16, t16);
    gemm(t16, wo16 + (size_t)l * D * D, enc_bo + l * D, t32, nullptr, MROWS, D, D, 0);
    ln_residual_kernel<<<MROWS, 256, 0, stream>>>(x32, t32, enc_ln1_g + l * D,
                                                  enc_ln1_b + l * D, x16);
    gemm(x16, w1_16 + (size_t)l * D * D, enc_b1 + l * D, nullptr, t16, MROWS, D, D, 1);
    gemm(t16, w2_16 + (size_t)l * D * D, enc_b2 + l * D, t32, nullptr, MROWS, D, D, 0);
    ln_residual_kernel<<<MROWS, 256, 0, stream>>>(x32, t32, enc_ln2_g + l * D,
                                                  enc_ln2_b + l * D, x16);
  }

  pool_kernel<<<N_NODES, 256, 0, stream>>>(x32, rule_types, type_emb, h32, h16);

  // ---- logical GNN ----------------------------------------------------------
  for (int l = 0; l < 4; ++l) {
    zero_kernel<<<(N_NODES * D + 255) / 256, 256, 0, stream>>>(hn32, N_NODES * D);
    for (int r = 0; r < 3; ++r) {
      gemm(h16, msg16 + (size_t)r * D * D, msg_b + r * D, nullptr, th16, N_NODES, D, D, 0);
      gemm(th16, gatw16 + (size_t)l * D * GH * D, nullptr, xg32, nullptr,
           N_NODES, GH * D, D, 0);
      gat_scores_kernel<<<N_NODES, 128, 0, stream>>>(xg32, gat_asrc + l * GH * D,
                                                     gat_adst + l * GH * D, a_s, a_d);
      gat_init_m_kernel<<<(N_NODES * GH + 255) / 256, 256, 0, stream>>>(a_s, a_d, eself,
                                                                        mbuf, N_NODES * GH);
      gat_edge_max_kernel<<<(NEDGES + 255) / 256, 256, 0, stream>>>(e_src, e_dst, edge_type,
                                                                    r, a_s, a_d, mbuf, NEDGES);
      zero_kernel<<<(N_NODES * GH + 255) / 256, 256, 0, stream>>>(denom, N_NODES * GH);
      zero_kernel<<<(N_NODES * GH * D + 255) / 256, 256, 0, stream>>>(num32, N_NODES * GH * D);
      gat_edge_accum_kernel<<<NEDGES, 128, 0, stream>>>(e_src, e_dst, edge_type, r,
                                                        a_s, a_d, mbuf, xg32, denom, num32);
      gat_self_finish_kernel<<<N_NODES, 256, 0, stream>>>(eself, mbuf, denom, num32, xg32,
                                                          gat_b + l * D, hn32);
    }
    relu_residual_kernel<<<N_NODES, 256, 0, stream>>>(h32, hn32, h16);
  }

  // ---- MLP + Lorentz exp-map ------------------------------------------------
  gemm(h16, sp1_16, sp_b1, nullptr, hid16, N_NODES, HID, D, 1);
  gemm(hid16, sp2_16, sp_b2, st32, nullptr, N_NODES, D, HID, 0);
  tanh_scale_kernel<<<(N_NODES * D + 255) / 256, 256, 0, stream>>>(st32, log_c, N_NODES * D);
  lorentz_kernel<<<N_NODES, 256, 0, stream>>>(st32, (float*)d_out);
}
